// RCNet_21818433863867
// MI455X (gfx1250) — matrix operand order repaired and checked by reading, not compiled, not run on hardware
//
#include <hip/hip_runtime.h>
#include <hip/hip_bf16.h>
#include <math.h>

// ---- problem constants (match reference) ----
#define Bc   4
#define Nc   65536
#define Gc   64
#define G3c  (Gc*Gc*Gc)          // 262144
#define CF   32
#define HF   128
#define WF   128
#define VSc  0.05f
#define R2c  (1.25f*1.25f)
#define DANc 0.5f

// workspace layout
#define OCC_WORDS  (Bc*G3c/32)               // 32768 u32 = 128KB
#define OCC_BYTES  (OCC_WORDS*4)
#define FLAG_OFF   OCC_BYTES                 // 4 ints
#define ACC_OFF    (OCC_BYTES + 64)          // B*4 floats
#define ZERO_WORDS ((ACC_OFF + Bc*4*4 + 3)/4)

typedef __attribute__((ext_vector_type(16))) _Float16 v16h;
typedef __attribute__((ext_vector_type(8)))  float    v8f;

// -------------------- zero workspace --------------------
__global__ void k_zero(unsigned* __restrict__ w, int n) {
    int i = blockIdx.x * blockDim.x + threadIdx.x;
    if (i < n) w[i] = 0u;
}

// -------------------- per-point nearby "any" reduction --------------------
__device__ __forceinline__ float min_d2_to_anchors(float px, float py, float pz,
                                                   float ax, float ay, float az) {
    float dx = px - ax, dy = py - ay;
    float dxy = dx*dx + dy*dy;
    float dz0 = pz - (az - DANc);
    float dz1 = pz - az;
    float dz2 = pz - (az + DANc);
    float d0 = dxy + dz0*dz0, d1 = dxy + dz1*dz1, d2 = dxy + dz2*dz2;
    return fminf(d0, fminf(d1, d2));
}

__global__ void k_nearby(const float* __restrict__ points,
                         const float* __restrict__ pelvis,
                         int* __restrict__ anyNear) {
    int i = blockIdx.x * blockDim.x + threadIdx.x;   // B*N threads
    int b = i >> 16;
    const float* p = points + (size_t)i * 3;
    float ax = pelvis[b*3+0], ay = pelvis[b*3+1], az = pelvis[b*3+2];
    bool nearby = min_d2_to_anchors(p[0], p[1], p[2], ax, ay, az) <= R2c;
    unsigned long long m = __ballot(nearby);
    if ((threadIdx.x & 31) == 0 && m != 0ull) atomicOr(&anyNear[b], 1);
}

// -------------------- voxel occupancy --------------------
__global__ void k_vox(const float* __restrict__ points,
                      const float* __restrict__ pelvis,
                      const int* __restrict__ anyNear,
                      unsigned* __restrict__ occW) {
    int i = blockIdx.x * blockDim.x + threadIdx.x;   // B*N threads
    int b = i >> 16;
    const float* p = points + (size_t)i * 3;
    float px = p[0], py = p[1], pz = p[2];
    float ax = pelvis[b*3+0], ay = pelvis[b*3+1], az = pelvis[b*3+2];
    bool nearby = min_d2_to_anchors(px, py, pz, ax, ay, az) <= R2c;
    if (!anyNear[b]) nearby = true;                   // fallback: all True
    int vx = (int)floorf(px / VSc), vy = (int)floorf(py / VSc), vz = (int)floorf(pz / VSc);
    int v0x = (int)floorf(ax / VSc) - Gc/2;
    int v0y = (int)floorf(ay / VSc) - Gc/2;
    int v0z = (int)floorf(az / VSc) - Gc/2;
    int lx = vx - v0x, ly = vy - v0y, lz = vz - v0z;
    bool inb = (lx >= 0) & (lx < Gc) & (ly >= 0) & (ly < Gc) &
               (lz >= 0) & (lz < Gc) & nearby;
    if (inb) {
        int flat = lx * (Gc*Gc) + ly * Gc + lz;
        int gi = b * G3c + flat;
        atomicOr(&occW[gi >> 5], 1u << (gi & 31));
    }
}

// -------------------- main WMMA MLP kernel --------------------
__device__ __forceinline__ float fetch_f(const float* __restrict__ fc, int x, int y) {
    if (x < 0 || x > WF-1 || y < 0 || y > HF-1) return 0.f;
    return fc[y * WF + x];
}

__global__ __launch_bounds__(256) void k_mlp(
    const float* __restrict__ fmap, const float* __restrict__ pelvis,
    const float* __restrict__ Kmat, const float* __restrict__ bbx,
    const float* __restrict__ W1, const float* __restrict__ b1,
    const float* __restrict__ W2, const float* __restrict__ b2,
    float* __restrict__ out) {

    __shared__ _Float16 hbuf[8][16][72];   // per-wave 16x64 hidden tile (+pad)
    __shared__ float    obuf[8][16*12];    // per-wave 16x12 output staging

    const int lane = threadIdx.x & 31;
    const int wave = threadIdx.x >> 5;
    const int row  = lane & 15;            // this lane's voxel row (M) / column (N)
    const int hi   = lane >> 4;            // half-wave select
    const int tile = blockIdx.x * 8 + wave;
    const int b     = tile >> 14;          // G3/16 = 16384 tiles per batch
    const int gbase = (tile & 16383) << 4;

    // ---- B fragments: W1 (4 N-tiles x 2 K-steps), W2 (2 K-steps) ----
    // B layout: lane holds col N=lane%16; element e -> K = (lane>=16?16:0)+e (+32*kstep)
    v16h bw1[4][2];
    v16h bw2[2];
    const int klane = hi * 16;
    #pragma unroll
    for (int nt = 0; nt < 4; ++nt)
        #pragma unroll
        for (int ks = 0; ks < 2; ++ks)
            #pragma unroll
            for (int e = 0; e < 16; ++e) {
                int Kk = ks * 32 + klane + e;
                float w = (Kk < 35) ? W1[Kk * 64 + nt * 16 + row] : 0.f;
                bw1[nt][ks][e] = (_Float16)w;
            }
    #pragma unroll
    for (int ks = 0; ks < 2; ++ks)
        #pragma unroll
        for (int e = 0; e < 16; ++e) {
            int Kk = ks * 32 + klane + e;  // always < 64
            float w = (row < 12) ? W2[Kk * 12 + row] : 0.f;
            bw2[ks][e] = (_Float16)w;
        }

    // ---- per-voxel geometry (voxel g = gbase + row, duplicated in both half-waves) ----
    const int g  = gbase + row;
    const int gx = g >> 12, gy = (g >> 6) & 63, gz = g & 63;
    const float ax = pelvis[b*3+0], ay = pelvis[b*3+1], az = pelvis[b*3+2];
    const int v0x = (int)floorf(ax / VSc) - Gc/2;
    const int v0y = (int)floorf(ay / VSc) - Gc/2;
    const int v0z = (int)floorf(az / VSc) - Gc/2;
    const float cxv = (v0x + gx) * VSc + VSc * 0.5f;
    const float cyv = (v0y + gy) * VSc + VSc * 0.5f;
    const float czv = (v0z + gz) * VSc + VSc * 0.5f;
    const float fx = Kmat[b*9+0], fy = Kmat[b*9+4];
    const float cx = Kmat[b*9+2], cy = Kmat[b*9+5];
    const float bl = bbx[b*4+0], bt = bbx[b*4+1], br = bbx[b*4+2], bb = bbx[b*4+3];
    const float u  = cxv / czv * fx + cx;
    const float vp = cyv / czv * fy + cy;
    const float U  = (u  - bl) / (br - bl) * (float)(WF - 1);
    const float V  = (vp - bt) / (bb - bt) * (float)(HF - 1);
    const float x0f = floorf(U), y0f = floorf(V);
    const float wx = U - x0f, wy = V - y0f;
    const int x0 = (int)x0f, y0 = (int)y0f;
    const float w00 = (1.f-wx)*(1.f-wy), w10 = wx*(1.f-wy);
    const float w01 = (1.f-wx)*wy,       w11 = wx*wy;
    const float* fmb = fmap + (size_t)b * CF * HF * WF;

    auto feat = [&](int Kk) -> float {
        if (Kk >= 35) return 0.f;
        if (Kk == 0)  return ax - cxv;
        if (Kk == 1)  return ay - cyv;
        if (Kk == 2)  return az - czv;
        const float* fc = fmb + (Kk - 3) * HF * WF;
        return w00 * fetch_f(fc, x0,   y0)   + w10 * fetch_f(fc, x0+1, y0) +
               w01 * fetch_f(fc, x0,   y0+1) + w11 * fetch_f(fc, x0+1, y0+1);
    };

    // ---- A fragments (16x32 f16): element e -> K = e + (lane>=16?16:0) ----
    v16h a0, a1;
    #pragma unroll
    for (int e = 0; e < 16; ++e) {
        int kb = e + hi * 16;
        a0[e] = (_Float16)feat(kb);
        a1[e] = (_Float16)feat(kb + 32);
    }

    // ---- layer 1: h = relu(feat @ W1 + b1), N = 64 (4 tiles of 16) ----
    v8f acc[4];
    #pragma unroll
    for (int nt = 0; nt < 4; ++nt) {
        float bias = b1[nt * 16 + row];
        v8f c = {bias, bias, bias, bias, bias, bias, bias, bias};
        c = __builtin_amdgcn_wmma_f32_16x16x32_f16(false, a0, false, bw1[nt][0],
                                                   (short)0, c, false, false);
        c = __builtin_amdgcn_wmma_f32_16x16x32_f16(false, a1, false, bw1[nt][1],
                                                   (short)0, c, false, false);
        acc[nt] = c;
    }

    // relu + C-layout -> LDS 16x64 tile (row M, col n), wave-private
    #pragma unroll
    for (int nt = 0; nt < 4; ++nt)
        #pragma unroll
        for (int rr = 0; rr < 8; ++rr) {
            float h = acc[nt][rr];
            h = h > 0.f ? h : 0.f;
            hbuf[wave][rr + hi * 8][nt * 16 + row] = (_Float16)h;
        }
    asm volatile("s_wait_dscnt 0" ::: "memory");   // same-wave LDS RAW

    // re-load hidden in A layout
    v16h a2_0, a2_1;
    #pragma unroll
    for (int e = 0; e < 16; ++e) {
        int kb = e + hi * 16;
        a2_0[e] = hbuf[wave][row][kb];
        a2_1[e] = hbuf[wave][row][kb + 32];
    }

    // ---- layer 2: out = h @ W2 + b2, N = 12 (one padded tile of 16) ----
    float bias2 = (row < 12) ? b2[row] : 0.f;
    v8f c2 = {bias2, bias2, bias2, bias2, bias2, bias2, bias2, bias2};
    c2 = __builtin_amdgcn_wmma_f32_16x16x32_f16(false, a2_0, false, bw2[0],
                                                (short)0, c2, false, false);
    c2 = __builtin_amdgcn_wmma_f32_16x16x32_f16(false, a2_1, false, bw2[1],
                                                (short)0, c2, false, false);

    // ---- stage 16x12 tile to LDS, then fully coalesced global store ----
    if (row < 12) {
        #pragma unroll
        for (int rr = 0; rr < 8; ++rr) {
            int M = rr + hi * 8;
            obuf[wave][M * 12 + row] = c2[rr];
        }
    }
    asm volatile("s_wait_dscnt 0" ::: "memory");   // same-wave LDS RAW

    float* obase = out + (size_t)(b * G3c + gbase) * 12;   // 192 contiguous floats
    #pragma unroll
    for (int q = 0; q < 6; ++q) {
        int idx = q * 32 + lane;                   // 128B/line per iteration
        obase[idx] = obuf[wave][idx];
    }
}

// -------------------- exp-weighted centroid accumulation --------------------
__global__ __launch_bounds__(256) void k_accum(
    const float* __restrict__ out, const unsigned* __restrict__ occW,
    const float* __restrict__ pelvis, float* __restrict__ accum) {
    int i = blockIdx.x * blockDim.x + threadIdx.x;   // B*G3 threads
    int b = i >> 18;
    int g = i & (G3c - 1);
    // out row is 48B (3x16B) -> 16B-aligned float4 covering [conf, ox, oy, oz]
    const float4 o4 = *(const float4*)(out + (size_t)i * 12);
    float conf = o4.x, ox = o4.y, oy = o4.z, oz = o4.w;
    bool occ = (occW[i >> 5] >> (i & 31)) & 1u;
    float e = 0.f;
    if (occ) {
        float s = 1.f / (1.f + expf(-conf));   // sigmoid in (0,1)
        e = expf(s);                           // shift-free softmax weight
    }
    int gx = g >> 12, gy = (g >> 6) & 63, gz = g & 63;
    float ax = pelvis[b*3+0], ay = pelvis[b*3+1], az = pelvis[b*3+2];
    int v0x = (int)floorf(ax / VSc) - Gc/2;
    int v0y = (int)floorf(ay / VSc) - Gc/2;
    int v0z = (int)floorf(az / VSc) - Gc/2;
    float cxv = (v0x + gx) * VSc + VSc * 0.5f;
    float cyv = (v0y + gy) * VSc + VSc * 0.5f;
    float czv = (v0z + gz) * VSc + VSc * 0.5f;
    float sx = e * (cxv + ox), sy = e * (cyv + oy), sz = e * (czv + oz);
    #pragma unroll
    for (int off = 16; off > 0; off >>= 1) {
        e  += __shfl_down(e,  off);
        sx += __shfl_down(sx, off);
        sy += __shfl_down(sy, off);
        sz += __shfl_down(sz, off);
    }
    if ((threadIdx.x & 31) == 0) {
        atomicAdd(&accum[b*4+0], e);
        atomicAdd(&accum[b*4+1], sx);
        atomicAdd(&accum[b*4+2], sy);
        atomicAdd(&accum[b*4+3], sz);
    }
}

// -------------------- finalize refined --------------------
__global__ void k_final(const float* __restrict__ pelvis,
                        const float* __restrict__ accum,
                        float* __restrict__ refined) {
    int b = threadIdx.x;
    if (b < Bc) {
        float se = accum[b*4+0];
        float x = accum[b*4+1] / se;
        float y = accum[b*4+2] / se;
        float z = accum[b*4+3] / se;
        bool bad = !(se > 0.f) || (x != x) || (y != y) || (z != z);
        refined[b*3+0] = bad ? pelvis[b*3+0] : x;
        refined[b*3+1] = bad ? pelvis[b*3+1] : y;
        refined[b*3+2] = bad ? pelvis[b*3+2] : z;
    }
}

extern "C" void kernel_launch(void* const* d_in, const int* in_sizes, int n_in,
                              void* d_out, int out_size, void* d_ws, size_t ws_size,
                              hipStream_t stream) {
    (void)in_sizes; (void)n_in; (void)out_size; (void)ws_size;
    const float* points = (const float*)d_in[0];
    const float* fmap   = (const float*)d_in[1];
    const float* pelvis = (const float*)d_in[2];
    const float* Kmat   = (const float*)d_in[3];
    const float* bbx    = (const float*)d_in[4];
    const float* W1     = (const float*)d_in[5];
    const float* b1     = (const float*)d_in[6];
    const float* W2     = (const float*)d_in[7];
    const float* b2     = (const float*)d_in[8];

    float* refined = (float*)d_out;            // (B,3) first in return order
    float* out     = refined + Bc * 3;         // (B,G3,12) second

    unsigned* occW    = (unsigned*)d_ws;
    int*      anyNear = (int*)((char*)d_ws + FLAG_OFF);
    float*    accum   = (float*)((char*)d_ws + ACC_OFF);

    k_zero  <<<(ZERO_WORDS + 255) / 256, 256, 0, stream>>>((unsigned*)d_ws, ZERO_WORDS);
    k_nearby<<<(Bc * Nc) / 256, 256, 0, stream>>>(points, pelvis, anyNear);
    k_vox   <<<(Bc * Nc) / 256, 256, 0, stream>>>(points, pelvis, anyNear, occW);
    k_mlp   <<<(Bc * G3c / 16) / 8, 256, 0, stream>>>(fmap, pelvis, Kmat, bbx,
                                                      W1, b1, W2, b2, out);
    k_accum <<<(Bc * G3c) / 256, 256, 0, stream>>>(out, occW, pelvis, accum);
    k_final <<<1, 32, 0, stream>>>(pelvis, accum, refined);
}
